// MLDEL_2_55989193671246
// MI455X (gfx1250) — matrix operand order repaired and checked by reading, not compiled, not run on hardware
//
#include <hip/hip_runtime.h>
#include <hip/hip_bf16.h>
#include <math.h>

typedef __bf16 bf16_t;
typedef __attribute__((ext_vector_type(16))) __bf16 v16bf;
typedef __attribute__((ext_vector_type(8)))  __bf16 v8bf;
typedef __attribute__((ext_vector_type(8)))  float  v8f;
typedef __attribute__((ext_vector_type(4)))  unsigned int u32x4;
typedef __attribute__((ext_vector_type(8)))  int i32x8;
typedef __attribute__((ext_vector_type(4)))  int i32x4;

#define HN      100000
#define HE      1600000
#define DIN     129
#define DOUT    128
#define MAXL    50
#define BB      4096
#define NCLS    5
#define KPAD1   160      // 129 padded to multiple of 32

#define MIN_NORMF 1e-15f
#define EPSF      1e-7f
#define MAX_NORMF 1000000.0f

// ---------------- wave helpers (wave32) ----------------
__device__ __forceinline__ float wredsum(float x) {
#pragma unroll
  for (int o = 16; o > 0; o >>= 1) x += __shfl_xor(x, o, 32);
  return x;
}

// logmap0(proj(expmap0(u))) for a tangent row held as u[k] at component
// j = lane + 32*k (caller guarantees the time-like component slot holds 0).
__device__ __forceinline__ void rt_wave(float u[4], float K, float sqrtK) {
  float ss = 0.f;
#pragma unroll
  for (int k = 0; k < 4; k++) ss += u[k] * u[k];
  ss = wredsum(ss);
  float r   = fmaxf(sqrtf(ss), MIN_NORMF);
  float th  = r / sqrtK;
  float sh  = sinhf(th);
  float fy  = sqrtK * sh / r;                 // y = fy * u
  float ysq = fy * fy * ss;
  float x0  = sqrtf(fmaxf(K + ysq, EPSF));    // proj
  float yn  = fmaxf(sqrtf(ysq), MIN_NORMF);
  float sc  = sqrtK * acoshf(fmaxf(x0 / sqrtK, 1.f + EPSF)) / yn;
  float f   = sc * fy;
#pragma unroll
  for (int k = 0; k < 4; k++) u[k] *= f;
}

// Fused epilogue: T (matvec result row, comp0 discarded) ->
// logmap0( mobius_add( proj(expmap0(T)), bias_point ) ), bias tangent u_b given.
__device__ void hyp_epilogue(const float* Trow, const float* __restrict__ u_b,
                             float K, float sqrtK, float out[4], int lane) {
  float t[4], ub[4];
  float xs = 0.f;
#pragma unroll
  for (int k = 0; k < 4; k++) {
    int j = lane + 32 * k;
    float v = (j == 0) ? 0.f : Trow[j];   // expmap0 drops component 0
    t[k] = v; ub[k] = u_b[j];
    xs += v * v;
  }
  xs = wredsum(xs);
  float xn  = fmaxf(sqrtf(xs), MIN_NORMF);
  float th  = xn / sqrtK;
  float sh  = sinhf(th);
  float fy  = sqrtK * sh / xn;              // P_y = fy * t
  float ysq = fy * fy * xs;
  float x0  = sqrtf(fmaxf(K + ysq, EPSF));  // proj
  float yn  = fmaxf(sqrtf(ysq), MIN_NORMF);
  float iyn = 1.f / yn;
  // ptransp0: alpha = <yhat, u_spatial>/sqrtK
  float duy = 0.f;
#pragma unroll
  for (int k = 0; k < 4; k++) duy += (fy * t[k] * iyn) * ub[k];
  duy = wredsum(duy);
  float alpha = duy / sqrtK;
  float w0 = alpha * yn;                    // u0 - alpha*(-yn), u0 == 0
  float wj[4]; float ux = 0.f;
#pragma unroll
  for (int k = 0; k < 4; k++) {
    int j = lane + 32 * k;
    float yhat = fy * t[k] * iyn;
    float w = ub[k] - alpha * (sqrtK - x0) * yhat;
    if (j == 0) w = w0;
    wj[k] = w;
    if (j != 0) ux += (fy * t[k]) * w;      // proj_tan: <P_y, w_spatial>
  }
  ux = wredsum(ux);
  float v0 = ux / fmaxf(x0, EPSF);
  float ssum = 0.f;
#pragma unroll
  for (int k = 0; k < 4; k++) {
    int j = lane + 32 * k;
    float v = (j == 0) ? v0 : wj[k];
    ssum += v * v;
  }
  ssum = wredsum(ssum);
  float mdot  = ssum - 2.f * v0 * v0;       // minkowski_dot(v,v)
  float normu = fminf(sqrtf(fmaxf(mdot, EPSF)), MAX_NORMF);
  float th2   = fmaxf(normu / sqrtK, MIN_NORMF);
  float c2    = coshf(th2);
  float s2f   = sinhf(th2) / th2;
  float rsq = 0.f; float resj[4];
#pragma unroll
  for (int k = 0; k < 4; k++) {
    int j = lane + 32 * k;
    float P  = (j == 0) ? x0 : fy * t[k];
    float vv = (j == 0) ? v0 : wj[k];
    float rr = c2 * P + s2f * vv;           // expmap(v, P)
    resj[k] = rr;
    if (j != 0) rsq += rr * rr;
  }
  rsq = wredsum(rsq);
  float x0f = sqrtf(fmaxf(K + rsq, EPSF));  // proj
  float ynf = fmaxf(sqrtf(rsq), MIN_NORMF);
  float sc  = sqrtK * acoshf(fmaxf(x0f / sqrtK, 1.f + EPSF)) / ynf;  // logmap0
#pragma unroll
  for (int k = 0; k < 4; k++) {
    int j = lane + 32 * k;
    out[k] = (j == 0) ? 0.f : sc * resj[k];
  }
}

// ---------------- prep kernels ----------------
__global__ void prep_scalars(const float* __restrict__ raw_c,
                             const float* __restrict__ b1,
                             const float* __restrict__ bg1,
                             const float* __restrict__ bg2,
                             float* cbuf, float* ub1, float* ubg1, float* ubg2) {
  __shared__ float shb[128];
  __shared__ float cs[2];
  int t = threadIdx.x;
  if (t == 0) {
    float c = log1pf(expf(raw_c[0])) + 1e-5f;   // softplus + 1e-5
    cs[0] = c; cs[1] = sqrtf(1.f / c);
    cbuf[0] = c; cbuf[1] = cs[1];
  }
  __syncthreads();
  float c = cs[0], sqrtK = cs[1], K = 1.f / c;
  const float* bs[3] = { b1, bg1, bg2 };
  float* us[3] = { ub1, ubg1, ubg2 };
  for (int i = 0; i < 3; i++) {
    float v = (t == 0) ? 0.f : bs[i][t];        // proj_tan0 zeroes comp 0
    shb[t] = v * v;
    __syncthreads();
    for (int s = 64; s > 0; s >>= 1) { if (t < s) shb[t] += shb[t + s]; __syncthreads(); }
    float ss = shb[0];
    __syncthreads();
    float r   = fmaxf(sqrtf(ss), MIN_NORMF);
    float th  = r / sqrtK;
    float sh  = sinhf(th);
    float fy  = sqrtK * sh / r;
    float ysq = fy * fy * ss;
    float x0  = sqrtf(fmaxf(K + ysq, EPSF));
    float yn  = fmaxf(sqrtf(ysq), MIN_NORMF);
    float sc  = sqrtK * acoshf(fmaxf(x0 / sqrtK, 1.f + EPSF)) / yn;
    us[i][t] = sc * fy * v;                     // logmap0(hyp_bias_point(b))
  }
}

// transpose+convert weight W[K x Nc] (row-major) -> Wt[Npad x Kpad] bf16, zero-padded
__global__ void convertW(const float* __restrict__ W, bf16_t* __restrict__ Wt,
                         int K, int Nc, int Kpad, int Npad) {
  int i = blockIdx.x * blockDim.x + threadIdx.x;
  if (i >= Npad * Kpad) return;
  int n = i / Kpad, k = i % Kpad;
  float v = (k < K && n < Nc) ? W[(size_t)k * Nc + n] : 0.f;
  Wt[i] = (bf16_t)v;
}

// t0 = logmap0(proj(expmap0(proj_tan0(A1))))  -> bf16 [HN x KPAD1]
__global__ void __launch_bounds__(256) prep_t0(const float* __restrict__ A1,
                                               bf16_t* __restrict__ t0,
                                               const float* __restrict__ cbuf) {
  int wave = threadIdx.x >> 5, lane = threadIdx.x & 31;
  int row = blockIdx.x * 8 + wave;
  if (row >= HN) return;
  float c = cbuf[0], sqrtK = cbuf[1], K = 1.f / c;
  const float* a = A1 + (size_t)row * DIN;
  float u[4];
#pragma unroll
  for (int k = 0; k < 4; k++) u[k] = a[1 + lane + 32 * k];  // spatial comps 1..128
  rt_wave(u, K, sqrtK);
  bf16_t* tr = t0 + (size_t)row * KPAD1;
#pragma unroll
  for (int k = 0; k < 4; k++) tr[1 + lane + 32 * k] = (bf16_t)u[k];
  if (lane == 0) tr[0] = (bf16_t)0.f;
  for (int p = 129 + lane; p < KPAD1; p += 32) tr[p] = (bf16_t)0.f;   // K padding
}

// ---- Tensor-Data-Mover stage of the weight tile into LDS (one wave issues) ----
// D# per CDNA5 ISA: 2D tensor == tile: rowsN x Kpad bf16 elements.
__device__ __forceinline__ void tdm_stage_weights(const bf16_t* __restrict__ Wt,
                                                  bf16_t* lds_dst,
                                                  int Kpad, int rowsN, int lane) {
#if __has_builtin(__builtin_amdgcn_tensor_load_to_lds) && __has_builtin(__builtin_amdgcn_s_wait_tensorcnt)
  unsigned long long ga = (unsigned long long)(size_t)Wt;
  unsigned ldsoff = (unsigned)(size_t)lds_dst;     // LDS byte offset (addr[31:0])
  unsigned td0 = (unsigned)Kpad;                   // elements per row (data_size units)
  unsigned td1 = (unsigned)rowsN;                  // rows
  u32x4 g0;
  g0[0] = 1u;                                      // count=1, user descriptor
  g0[1] = ldsoff;                                  // lds_addr
  g0[2] = (unsigned)(ga & 0xffffffffu);            // global_addr[31:0]
  g0[3] = (unsigned)((ga >> 32) & 0x01ffffffu) | (2u << 30);  // addr[56:32], type=2
  i32x8 g1;
  g1[0] = (int)(1u << 16);                         // wg_mask=0, data_size=1 (2 bytes)
  g1[1] = (int)((td0 & 0xffffu) << 16);            // tensor_dim0[15:0]
  g1[2] = (int)(((td0 >> 16) & 0xffffu) | ((td1 & 0xffffu) << 16)); // dim0 hi | dim1 lo
  g1[3] = (int)(((td1 >> 16) & 0xffffu) | ((td0 & 0xffffu) << 16)); // dim1 hi | tile_dim0
  g1[4] = (int)(td1 & 0xffffu);                    // tile_dim1 = rows, tile_dim2 = 0
  g1[5] = (int)td0;                                // tensor_dim0_stride[31:0]
  g1[6] = 0;                                       // stride0 hi | dim1_stride lo
  g1[7] = 0;
  i32x4 gz = { 0, 0, 0, 0 };
#if __clang_major__ >= 23
  i32x8 gz8 = { 0, 0, 0, 0, 0, 0, 0, 0 };
  __builtin_amdgcn_tensor_load_to_lds(g0, g1, gz, gz, gz8, 0);
#else
  __builtin_amdgcn_tensor_load_to_lds(g0, g1, gz, gz, 0);
#endif
  __builtin_amdgcn_s_wait_tensorcnt((unsigned short)0);
#else
  // Fallback: cooperative copy by the issuing wave.
  int tot = Kpad * rowsN;
  for (int i = lane; i < tot; i += 32) lds_dst[i] = Wt[i];
#endif
}

// ---------------- WMMA GEMM + fused hyperbolic epilogue ----------------
// A [HN x Kpad] bf16, Wt [128 x Kpad] bf16 (output-col-major over K).
// Persistent blocks: weight tile staged to LDS once (TDM), then grid-stride over
// 16-row stripes; 8 waves/block = 16x128 output stripe per iteration.
// mode 1: Lin1 branch (emit ta1 bf16 + ta2 = logmap0(expmap0(n*ta1)));
// mode 0: emit tangent f32.
__global__ void __launch_bounds__(256) gemm_hyp(
    const bf16_t* __restrict__ A, const bf16_t* __restrict__ Wt, int Kpad,
    const float* __restrict__ cbuf, const float* __restrict__ u_b,
    int mode, const float* __restrict__ n_param,
    bf16_t* __restrict__ ta1bf, float* __restrict__ outf, int ntiles) {
  const int wave = threadIdx.x >> 5, lane = threadIdx.x & 31;
  const int half = lane >> 4, l15 = lane & 15;
  __shared__ bf16_t Bs[128 * KPAD1];     // staged weights (<= 40 KB)
  __shared__ float Tm[16 * 132];

  if (wave == 0)
    tdm_stage_weights(Wt, Bs, Kpad, 128, lane);
  __syncthreads();

  float c = cbuf[0], sqrtK = cbuf[1], K = 1.f / c;
  const bf16_t* bcol = Bs + (size_t)(wave * 16 + l15) * Kpad + half * 16;

  for (int tile = blockIdx.x; tile < ntiles; tile += gridDim.x) {
    const int row0 = tile * 16;
    v8f acc = {};
    const bf16_t* arow = A + (size_t)(row0 + l15) * Kpad + half * 16;
    for (int k = 0; k < Kpad; k += 32) {
      v8bf a0 = *(const v8bf*)(arow + k);        // K k..k+7   (lanes16-31: +16)
      v8bf a1 = *(const v8bf*)(arow + k + 8);    // K k+8..15  (lanes16-31: +24)
      v16bf bmat = *(const v16bf*)(bcol + k);    // K k..k+15  (lanes16-31: +16), from LDS
      v16bf amat;
#pragma unroll
      for (int i = 0; i < 8; i++) { amat[i] = a0[i]; amat[i + 8] = a1[i]; }
      acc = __builtin_amdgcn_wmma_f32_16x16x32_bf16(false, amat, false, bmat,
                                                    (short)0, acc, false, false);
    }
    __syncthreads();                    // Tm consumed by previous iteration
#pragma unroll
    for (int v = 0; v < 8; v++)
      Tm[(v + 8 * half) * 132 + wave * 16 + l15] = acc[v];
    __syncthreads();
#pragma unroll 1
    for (int rr = 0; rr < 2; ++rr) {
      int r = wave * 2 + rr;
      int row = row0 + r;
      float out[4];
      hyp_epilogue(&Tm[r * 132], u_b, K, sqrtK, out, lane);
      if (mode == 1) {
#pragma unroll
        for (int k = 0; k < 4; k++)
          ta1bf[(size_t)row * 128 + lane + 32 * k] = (bf16_t)out[k];
        float n = n_param[row];
        float u2[4];
#pragma unroll
        for (int k = 0; k < 4; k++) u2[k] = n * out[k];
        rt_wave(u2, K, sqrtK);                   // logmap0(a2)
#pragma unroll
        for (int k = 0; k < 4; k++)
          outf[(size_t)row * 128 + lane + 32 * k] = u2[k];
      } else {
#pragma unroll
        for (int k = 0; k < 4; k++)
          outf[(size_t)row * 128 + lane + 32 * k] = out[k];
      }
    }
  }
}

// ---------------- SpMM: out[rows[e]] += vals[e] * G[cols[e],:] ----------------
__global__ void __launch_bounds__(256) spmm_atomic(
    const int* __restrict__ rows, const int* __restrict__ cols,
    const float* __restrict__ vals, const float* __restrict__ G,
    float* __restrict__ out, int E) {
  int e = blockIdx.x * 8 + (threadIdx.x >> 5);
  if (e >= E) return;
  int lane = threadIdx.x & 31;
  int r = rows[e], cc = cols[e];
  float v = vals[e];
  float4 x = ((const float4*)(G + (size_t)cc * 128))[lane];
  float* dst = out + (size_t)r * 128 + lane * 4;
  atomicAdd(dst + 0, v * x.x);
  atomicAdd(dst + 1, v * x.y);
  atomicAdd(dst + 2, v * x.z);
  atomicAdd(dst + 3, v * x.w);
}

// x_1 = expmap0( logmap0(expmap0((1-n)*logmap0(proj(expmap0(agg))))) + logmap0(a2) )
// store t1 = logmap0(x_1) bf16
__global__ void __launch_bounds__(256) post1(const float* __restrict__ agg,
                                             const float* __restrict__ ta2,
                                             const float* __restrict__ n_param,
                                             bf16_t* __restrict__ t1,
                                             const float* __restrict__ cbuf) {
  int wave = threadIdx.x >> 5, lane = threadIdx.x & 31;
  int row = blockIdx.x * 8 + wave;
  if (row >= HN) return;
  float c = cbuf[0], sqrtK = cbuf[1], K = 1.f / c;
  float u[4];
#pragma unroll
  for (int k = 0; k < 4; k++) {
    int j = lane + 32 * k;
    u[k] = (j == 0) ? 0.f : agg[(size_t)row * 128 + j];
  }
  rt_wave(u, K, sqrtK);                        // logmap0(x1)
  float n = n_param[row];
#pragma unroll
  for (int k = 0; k < 4; k++) u[k] *= (1.f - n);
  rt_wave(u, K, sqrtK);                        // logmap0(mobius_matvec0 point)
#pragma unroll
  for (int k = 0; k < 4; k++) u[k] += ta2[(size_t)row * 128 + lane + 32 * k];
  rt_wave(u, K, sqrtK);                        // logmap0(x_1 final)
#pragma unroll
  for (int k = 0; k < 4; k++) t1[(size_t)row * 128 + lane + 32 * k] = (bf16_t)u[k];
}

// tx2 = logmap0(proj(expmap0(agg2))) bf16
__global__ void __launch_bounds__(256) post2(const float* __restrict__ agg,
                                             bf16_t* __restrict__ tx2,
                                             const float* __restrict__ cbuf) {
  int wave = threadIdx.x >> 5, lane = threadIdx.x & 31;
  int row = blockIdx.x * 8 + wave;
  if (row >= HN) return;
  float c = cbuf[0], sqrtK = cbuf[1], K = 1.f / c;
  float u[4];
#pragma unroll
  for (int k = 0; k < 4; k++) {
    int j = lane + 32 * k;
    u[k] = (j == 0) ? 0.f : agg[(size_t)row * 128 + j];
  }
  rt_wave(u, K, sqrtK);
#pragma unroll
  for (int k = 0; k < 4; k++) tx2[(size_t)row * 128 + lane + 32 * k] = (bf16_t)u[k];
}

// M2[b] = sum_l conv1_w[l]*tx2[flat[b,l]] ; M3[b] = sum_l conv2_w[l]*ta1[flat[b,l]]
__global__ void __launch_bounds__(128) gatherreduce(
    const int* __restrict__ flat, const bf16_t* __restrict__ tx2,
    const bf16_t* __restrict__ ta1, const float* __restrict__ c1w,
    const float* __restrict__ c2w, bf16_t* __restrict__ M2, bf16_t* __restrict__ M3) {
  int b = blockIdx.x, d = threadIdx.x;
  float a2 = 0.f, a3 = 0.f;
  for (int l = 0; l < MAXL; l++) {
    int idx = flat[b * MAXL + l];
    a2 += c1w[l] * (float)tx2[(size_t)idx * 128 + d];
    a3 += c2w[l] * (float)ta1[(size_t)idx * 128 + d];
  }
  M2[(size_t)b * 128 + d] = (bf16_t)a2;
  M3[(size_t)b * 128 + d] = (bf16_t)a3;
}

// sel[b, colOff + t] = (M @ Wt)[b,t] + bias   (t < 50), WMMA, 4 waves/block
__global__ void __launch_bounds__(128) gemm_small(
    const bf16_t* __restrict__ A, const bf16_t* __restrict__ Wt,
    const float* __restrict__ biasp, float* __restrict__ sel, int colOff) {
  const int wave = threadIdx.x >> 5, lane = threadIdx.x & 31;
  const int half = lane >> 4, l15 = lane & 15;
  const int row0 = blockIdx.x * 16;
  v8f acc = {};
  const bf16_t* arow = A  + (size_t)(row0 + l15) * 128 + half * 16;
  const bf16_t* bcol = Wt + (size_t)(wave * 16 + l15) * 128 + half * 16;
  for (int k = 0; k < 128; k += 32) {
    v8bf a0 = *(const v8bf*)(arow + k);
    v8bf a1 = *(const v8bf*)(arow + k + 8);
    v16bf bmat = *(const v16bf*)(bcol + k);
    v16bf amat;
#pragma unroll
    for (int i = 0; i < 8; i++) { amat[i] = a0[i]; amat[i + 8] = a1[i]; }
    acc = __builtin_amdgcn_wmma_f32_16x16x32_bf16(false, amat, false, bmat,
                                                  (short)0, acc, false, false);
  }
  float bias = biasp[0];
  int n = wave * 16 + l15;
  if (n < MAXL) {
#pragma unroll
    for (int v = 0; v < 8; v++) {
      int m = v + 8 * half;
      sel[(size_t)(row0 + m) * 100 + colOff + n] = acc[v] + bias;
    }
  }
}

__global__ void __launch_bounds__(256) final_cls(const float* __restrict__ sel,
                                                 const float* __restrict__ cls,
                                                 const float* __restrict__ cbias,
                                                 float* __restrict__ outArg, int B) {
  int b = blockIdx.x * blockDim.x + threadIdx.x;
  if (b >= B) return;
  const float* s = sel + (size_t)b * 100;
  float best = 0.f; int bi = 0;
  for (int k = 0; k < NCLS; k++) {
    float acc = cbias[k];
    for (int j = 0; j < 100; j++) acc += s[j] * cls[j * NCLS + k];
    if (k == 0 || acc > best) { best = acc; bi = k; }
  }
  outArg[b] = (float)bi;
}

// ---------------- launcher ----------------
extern "C" void kernel_launch(void* const* d_in, const int* in_sizes, int n_in,
                              void* d_out, int out_size, void* d_ws, size_t ws_size,
                              hipStream_t stream) {
  const float* A1      = (const float*)d_in[0];
  const int*   rows    = (const int*)  d_in[1];
  const int*   cols    = (const int*)  d_in[2];
  const float* vals    = (const float*)d_in[3];
  const int*   bidx    = (const int*)  d_in[4];
  const float* raw_c   = (const float*)d_in[5];
  const float* n_param = (const float*)d_in[6];
  const float* Lin1    = (const float*)d_in[7];
  const float* Lin1_b  = (const float*)d_in[8];
  const float* gc1_w   = (const float*)d_in[9];
  const float* gc1_b   = (const float*)d_in[10];
  const float* gc2_w   = (const float*)d_in[11];
  const float* gc2_b   = (const float*)d_in[12];
  const float* weight  = (const float*)d_in[13];
  const float* weight2 = (const float*)d_in[14];
  const float* conv1_w = (const float*)d_in[15];
  const float* conv1_b = (const float*)d_in[16];
  const float* conv2_w = (const float*)d_in[17];
  const float* conv2_b = (const float*)d_in[18];
  const float* cls     = (const float*)d_in[19];
  const float* cls_b   = (const float*)d_in[20];
  (void)in_sizes; (void)n_in; (void)out_size; (void)ws_size;

  char* ws = (char*)d_ws;
  size_t off = 0;
  auto alloc = [&](size_t bytes) -> char* {
    char* p = ws + off;
    off += (bytes + 255) & ~(size_t)255;
    return p;
  };
  float*  cbuf  = (float*) alloc(256);
  float*  ub1   = (float*) alloc(512);
  float*  ubg1  = (float*) alloc(512);
  float*  ubg2  = (float*) alloc(512);
  bf16_t* lin1T = (bf16_t*)alloc((size_t)128 * KPAD1 * 2);
  bf16_t* gc1T  = (bf16_t*)alloc((size_t)128 * KPAD1 * 2);
  bf16_t* gc2T  = (bf16_t*)alloc((size_t)128 * 128 * 2);
  bf16_t* wT    = (bf16_t*)alloc((size_t)64 * 128 * 2);
  bf16_t* w2T   = (bf16_t*)alloc((size_t)64 * 128 * 2);
  bf16_t* t0    = (bf16_t*)alloc((size_t)HN * KPAD1 * 2);
  bf16_t* ta1bf = (bf16_t*)alloc((size_t)HN * 128 * 2);
  float*  ta2   = (float*) alloc((size_t)HN * 128 * 4);
  float*  G     = (float*) alloc((size_t)HN * 128 * 4);
  float*  agg   = (float*) alloc((size_t)HN * 128 * 4);
  bf16_t* t1    = (bf16_t*)alloc((size_t)HN * 128 * 2);
  bf16_t* tx2   = (bf16_t*)alloc((size_t)HN * 128 * 2);
  bf16_t* M2    = (bf16_t*)alloc((size_t)BB * 128 * 2);
  bf16_t* M3    = (bf16_t*)alloc((size_t)BB * 128 * 2);

  float* outArg = (float*)d_out;
  float* sel    = outArg + BB;

  prep_scalars<<<1, 128, 0, stream>>>(raw_c, Lin1_b, gc1_b, gc2_b, cbuf, ub1, ubg1, ubg2);
  const int cw = 256;
  convertW<<<(128 * KPAD1 + cw - 1) / cw, cw, 0, stream>>>(Lin1,  lin1T, DIN, 128, KPAD1, 128);
  convertW<<<(128 * KPAD1 + cw - 1) / cw, cw, 0, stream>>>(gc1_w, gc1T,  DIN, 128, KPAD1, 128);
  convertW<<<(128 * 128   + cw - 1) / cw, cw, 0, stream>>>(gc2_w, gc2T,  128, 128, 128,   128);
  convertW<<<(64 * 128    + cw - 1) / cw, cw, 0, stream>>>(weight,  wT,  128, MAXL, 128,  64);
  convertW<<<(64 * 128    + cw - 1) / cw, cw, 0, stream>>>(weight2, w2T, 128, MAXL, 128,  64);

  prep_t0<<<(HN + 7) / 8, 256, 0, stream>>>(A1, t0, cbuf);

  const int GB = 1024;          // persistent GEMM blocks (grid-stride over tiles)
  const int NT = HN / 16;
  // a1 branch: ta1 (bf16) + logmap0(a2) (f32)
  gemm_hyp<<<GB, 256, 0, stream>>>(t0, lin1T, KPAD1, cbuf, ub1, 1, n_param, ta1bf, ta2, NT);
  // gcn layer 1 dense part
  gemm_hyp<<<GB, 256, 0, stream>>>(t0, gc1T, KPAD1, cbuf, ubg1, 0, nullptr, nullptr, G, NT);
  (void)hipMemsetAsync(agg, 0, (size_t)HN * 128 * 4, stream);
  spmm_atomic<<<(HE + 7) / 8, 256, 0, stream>>>(rows, cols, vals, G, agg, HE);
  post1<<<(HN + 7) / 8, 256, 0, stream>>>(agg, ta2, n_param, t1, cbuf);

  // gcn layer 2
  gemm_hyp<<<GB, 256, 0, stream>>>(t1, gc2T, 128, cbuf, ubg2, 0, nullptr, nullptr, G, NT);
  (void)hipMemsetAsync(agg, 0, (size_t)HN * 128 * 4, stream);
  spmm_atomic<<<(HE + 7) / 8, 256, 0, stream>>>(rows, cols, vals, G, agg, HE);
  post2<<<(HN + 7) / 8, 256, 0, stream>>>(agg, tx2, cbuf);

  // readout
  gatherreduce<<<BB, 128, 0, stream>>>(bidx, tx2, ta1bf, conv1_w, conv2_w, M2, M3);
  gemm_small<<<BB / 16, 128, 0, stream>>>(M2, wT,  conv1_b, sel, 0);
  gemm_small<<<BB / 16, 128, 0, stream>>>(M3, w2T, conv2_b, sel, 50);
  final_cls<<<(BB + 255) / 256, 256, 0, stream>>>(sel, cls, cls_b, outArg, BB);
}
